// KoLeoLoss_80255758893850
// MI455X (gfx1250) — matrix operand order repaired and checked by reading, not compiled, not run on hardware
//
#include <hip/hip_runtime.h>
#include <math.h>

// ---------------------------------------------------------------------------
// KoLeo loss for X[16384, 768] fp32 on gfx1250 (MI455X, wave32, WMMA).
// Gram/argmax via v_wmma_f32_16x16x32_bf16, two-term bf16 split (hi+lo) for
// near-fp32 argmax fidelity. B-panel staged in LDS via async global->LDS DMA
// (double buffered), A streamed from global, 2x4 register blocking per wave.
// ---------------------------------------------------------------------------

#define N_ROWS 16384
#define DIM    768
#define EPSF   1e-8f

// LDS staging: 64 cols x 32 K bf16, hi+lo, double buffered, 80B row stride.
#define LDS_ROW_STRIDE 80
#define LDS_ARR  (64 * LDS_ROW_STRIDE)       // 5120 B (one of hi/lo)
#define LDS_BUF  (2 * LDS_ARR)               // 10240 B (hi+lo)
#define LDS_TOT  (2 * LDS_BUF)               // 20480 B (double buffer)

typedef __attribute__((ext_vector_type(16))) __bf16 v16bf;
typedef __attribute__((ext_vector_type(8)))  __bf16 v8bf;
typedef __attribute__((ext_vector_type(8)))  float  v8f;
typedef __attribute__((ext_vector_type(4)))  int    v4i;

// Builtin expects: (int4 AS1* gsrc, int4 AS3* ldst, imm offset, imm cpol).
typedef __attribute__((address_space(1))) v4i g_v4i;
typedef __attribute__((address_space(3))) v4i l_v4i;

union FragU { v16bf v; v8bf h[2]; };

// --------------------------- async copy plumbing ---------------------------
#if __has_builtin(__builtin_amdgcn_global_load_async_to_lds_b128)
#define HAVE_ASYNC 1
__device__ __forceinline__ void async_copy_b128(const void* g, void* l) {
  __builtin_amdgcn_global_load_async_to_lds_b128(
      (g_v4i*)(uintptr_t)g,
      (l_v4i*)(unsigned)(uintptr_t)l, 0, 0);
}
#else
#define HAVE_ASYNC 0
__device__ __forceinline__ void async_copy_b128(const void* g, void* l) {
  *(v8bf*)l = *(const v8bf*)g;   // fallback: global load + ds_store
}
#endif

__device__ __forceinline__ void wait_async0() {
#if __has_builtin(__builtin_amdgcn_s_wait_asynccnt)
  __builtin_amdgcn_s_wait_asynccnt(0);
#elif HAVE_ASYNC
  asm volatile("s_wait_asynccnt 0" ::: "memory");
#endif
}

// ---------------------------------------------------------------------------
// Kernel 1: row L2-normalize, split normalized value into bf16 hi + bf16 lo.
// ---------------------------------------------------------------------------
__global__ void __launch_bounds__(256)
normalize_split_kernel(const float* __restrict__ X,
                       __bf16* __restrict__ Xhi,
                       __bf16* __restrict__ Xlo,
                       float*  __restrict__ invn) {
  const int row = blockIdx.x;
  const int tid = threadIdx.x;
  const float* xr = X + (size_t)row * DIM;

  float x0 = xr[tid], x1 = xr[tid + 256], x2 = xr[tid + 512];
  float ss = x0 * x0 + x1 * x1 + x2 * x2;

  #pragma unroll
  for (int m = 16; m; m >>= 1) ss += __shfl_xor(ss, m, 32);

  __shared__ float part[8];
  __shared__ float sinv;
  if ((tid & 31) == 0) part[tid >> 5] = ss;
  __syncthreads();
  if (tid == 0) {
    float s = 0.f;
    #pragma unroll
    for (int w = 0; w < 8; ++w) s += part[w];
    sinv = 1.0f / fmaxf(sqrtf(s), EPSF);   // X / max(||x||, eps)
  }
  __syncthreads();
  const float inv = sinv;
  if (tid == 0) invn[row] = inv;

  float xs[3] = {x0, x1, x2};
  #pragma unroll
  for (int e = 0; e < 3; ++e) {
    float xn = xs[e] * inv;
    __bf16 h = (__bf16)xn;
    __bf16 l = (__bf16)(xn - (float)h);
    size_t o = (size_t)row * DIM + tid + e * 256;
    Xhi[o] = h;
    Xlo[o] = l;
  }
}

// ---------------------------------------------------------------------------
// Fragment loaders.
// A 16x32 (ISA table): lanes 0-15 -> M, K {kk..kk+7} in v0-3, {kk+16..kk+23}
// in v4-7; lanes 16-31 the +8 chunks.
// B 32x16 (same table as A): lanes 0-15 -> N, K {kk..kk+7} in v0-3,
// {kk+16..kk+23} in v4-7; lanes 16-31 the +8 chunks.
// ---------------------------------------------------------------------------
__device__ __forceinline__ v16bf load_fragA(const __bf16* __restrict__ p,
                                            int row0, int kk,
                                            int laneHalf, int hiHalf) {
  const __bf16* q = p + (size_t)(row0 + laneHalf) * DIM + kk + hiHalf * 8;
  FragU f;
  f.h[0] = *(const v8bf*)(q);
  f.h[1] = *(const v8bf*)(q + 16);
  return f.v;
}

// B fragment from LDS-staged 64x32 tile (tile-local rows, 80B row stride).
__device__ __forceinline__ v16bf lds_fragB(const char* __restrict__ shArr,
                                           int t, int laneHalf, int hiHalf) {
  const char* q = shArr + (16 * t + laneHalf) * LDS_ROW_STRIDE + hiHalf * 16;
  FragU f;
  f.h[0] = *(const v8bf*)(q);
  f.h[1] = *(const v8bf*)(q + 32);
  return f.v;
}

__device__ __forceinline__ v8f wmma_bf16(v16bf a, v16bf b, v8f c) {
  return __builtin_amdgcn_wmma_f32_16x16x32_bf16(
      false, a, false, b, (short)0, c, false, false);
}

__device__ __forceinline__ void argmax_update(v8f acc, int colBase,
                                              int laneHalf, int myRow0,
                                              float* __restrict__ best,
                                              int* __restrict__ bidx) {
  const int col = colBase + laneHalf;
  #pragma unroll
  for (int g = 0; g < 8; ++g) {
    float v = acc[g];
    const bool self = (col == myRow0 + g);   // diagonal mask
    if (!self && v > best[g]) { best[g] = v; bidx[g] = col; }
  }
}

// Stage the 64-col x 32-K (hi+lo) B panel slice owned by this wave into LDS.
// Wave w stages tile-local rows 16w..16w+15; 4 lanes cover each 64B row chunk.
__device__ __forceinline__ void stage_panel(const __bf16* __restrict__ Xhi,
                                            const __bf16* __restrict__ Xlo,
                                            char* __restrict__ shBuf,
                                            int c, int kk, int wave, int lane) {
  const int r  = lane >> 2;         // 0..7
  const int ch = (lane & 3) * 16;   // byte within 64B row chunk
  #pragma unroll
  for (int q = 0; q < 2; ++q) {
    const int row = 16 * wave + 8 * q + r;                 // 0..63
    const size_t ge = (size_t)(c + row) * DIM + kk;        // element index
    async_copy_b128((const char*)(Xhi + ge) + ch,
                    shBuf + row * LDS_ROW_STRIDE + ch);
    async_copy_b128((const char*)(Xlo + ge) + ch,
                    shBuf + LDS_ARR + row * LDS_ROW_STRIDE + ch);
  }
}

// ---------------------------------------------------------------------------
// Kernel 2: Gram + running argmax.
// Block = 4 waves = 128 rows; each wave owns 32 rows (MT=2 row tiles) and
// sweeps all columns in 64-wide panels (NT=4) staged in LDS, double buffered.
// dot = hi*hi + hi*lo + lo*hi  (3 WMMAs per 16x16 tile per K-step).
// ---------------------------------------------------------------------------
__global__ void __launch_bounds__(128)
gram_argmax_kernel(const __bf16* __restrict__ Xhi,
                   const __bf16* __restrict__ Xlo,
                   int* __restrict__ bestIdx) {
  __shared__ __align__(16) char sh[LDS_TOT];

  const int lane     = threadIdx.x & 31;
  const int laneHalf = lane & 15;
  const int hiHalf   = lane >> 4;
  const int wave     = threadIdx.x >> 5;                 // 0..3
  const int gwave    = blockIdx.x * 4 + wave;
  const int rowBase  = gwave * 32;                       // 32 rows per wave

  float best[2][8];
  int   bidx[2][8];
  #pragma unroll
  for (int mt = 0; mt < 2; ++mt)
    #pragma unroll
    for (int g = 0; g < 8; ++g) { best[mt][g] = -2.0f; bidx[mt][g] = 0; }

  const int myRow0[2] = { rowBase + hiHalf * 8, rowBase + 16 + hiHalf * 8 };

  // Prologue: stage first panel chunk.
  stage_panel(Xhi, Xlo, sh, /*c=*/0, /*kk=*/0, wave, lane);
  wait_async0();
  __syncthreads();

  int buf = 0;
  for (int c = 0; c < N_ROWS; c += 64) {
    v8f acc[2][4];
    #pragma unroll
    for (int mt = 0; mt < 2; ++mt)
      #pragma unroll
      for (int t = 0; t < 4; ++t) acc[mt][t] = (v8f){};

    for (int kk = 0; kk < DIM; kk += 32) {
      // Prefetch next (c,kk) chunk into the other buffer.
      int cn = c, kn = kk + 32;
      if (kn == DIM) { kn = 0; cn = c + 64; }
      if (cn < N_ROWS)
        stage_panel(Xhi, Xlo, sh + (buf ^ 1) * LDS_BUF, cn, kn, wave, lane);

      // A fragments for this wave's two 16-row tiles (global, L2-hot).
      const v16bf ahi0 = load_fragA(Xhi, rowBase,      kk, laneHalf, hiHalf);
      const v16bf alo0 = load_fragA(Xlo, rowBase,      kk, laneHalf, hiHalf);
      const v16bf ahi1 = load_fragA(Xhi, rowBase + 16, kk, laneHalf, hiHalf);
      const v16bf alo1 = load_fragA(Xlo, rowBase + 16, kk, laneHalf, hiHalf);

      const char* shHi = sh + buf * LDS_BUF;
      const char* shLo = shHi + LDS_ARR;

      #pragma unroll
      for (int t = 0; t < 4; ++t) {
        const v16bf bh = lds_fragB(shHi, t, laneHalf, hiHalf);
        const v16bf bl = lds_fragB(shLo, t, laneHalf, hiHalf);
        acc[0][t] = wmma_bf16(ahi0, bh, acc[0][t]);
        acc[0][t] = wmma_bf16(ahi0, bl, acc[0][t]);
        acc[0][t] = wmma_bf16(alo0, bh, acc[0][t]);
        acc[1][t] = wmma_bf16(ahi1, bh, acc[1][t]);
        acc[1][t] = wmma_bf16(ahi1, bl, acc[1][t]);
        acc[1][t] = wmma_bf16(alo1, bh, acc[1][t]);
      }

      wait_async0();     // my async writes to buf^1 are complete
      __syncthreads();   // everyone done reading buf / writes visible
      buf ^= 1;
    }

    #pragma unroll
    for (int mt = 0; mt < 2; ++mt) {
      argmax_update(acc[mt][0], c,      laneHalf, myRow0[mt], best[mt], bidx[mt]);
      argmax_update(acc[mt][1], c + 16, laneHalf, myRow0[mt], best[mt], bidx[mt]);
      argmax_update(acc[mt][2], c + 32, laneHalf, myRow0[mt], best[mt], bidx[mt]);
      argmax_update(acc[mt][3], c + 48, laneHalf, myRow0[mt], best[mt], bidx[mt]);
    }
  }

  // Reduce over the 16 lanes sharing each row (xor masks < 16 stay in-half).
  #pragma unroll
  for (int mt = 0; mt < 2; ++mt) {
    #pragma unroll
    for (int g = 0; g < 8; ++g) {
      float v = best[mt][g];
      int   ix = bidx[mt][g];
      #pragma unroll
      for (int m = 8; m; m >>= 1) {
        float v2 = __shfl_xor(v, m, 32);
        int   i2 = __shfl_xor(ix, m, 32);
        if (v2 > v || (v2 == v && i2 < ix)) { v = v2; ix = i2; }
      }
      if (laneHalf == 0) bestIdx[myRow0[mt] + g] = ix;
    }
  }
}

// ---------------------------------------------------------------------------
// Kernel 3: per-row fp32 distance to nearest neighbor + log. One block/row.
// ---------------------------------------------------------------------------
__global__ void __launch_bounds__(256)
logdist_kernel(const float* __restrict__ X,
               const float* __restrict__ invn,
               const int*   __restrict__ bestIdx,
               float* __restrict__ logd) {
  const int i   = blockIdx.x;
  const int tid = threadIdx.x;
  const int j   = bestIdx[i];
  const float invi = invn[i];
  const float invj = invn[j];
  const float* xi = X + (size_t)i * DIM;
  const float* xj = X + (size_t)j * DIM;

  float ss = 0.f;
  #pragma unroll
  for (int e = 0; e < 3; ++e) {
    const int k = tid + e * 256;
    const float d = xi[k] * invi - xj[k] * invj + EPSF;
    ss += d * d;
  }
  #pragma unroll
  for (int m = 16; m; m >>= 1) ss += __shfl_xor(ss, m, 32);

  __shared__ float part[8];
  if ((tid & 31) == 0) part[tid >> 5] = ss;
  __syncthreads();
  if (tid == 0) {
    float s = 0.f;
    #pragma unroll
    for (int w = 0; w < 8; ++w) s += part[w];
    logd[i] = logf(sqrtf(s) + EPSF);
  }
}

// ---------------------------------------------------------------------------
// Kernel 4: deterministic single-block reduction -> loss = -mean(logd).
// ---------------------------------------------------------------------------
__global__ void __launch_bounds__(256)
reduce_loss_kernel(const float* __restrict__ logd, float* __restrict__ out) {
  __shared__ float s[256];
  float a = 0.f;
  for (int i = threadIdx.x; i < N_ROWS; i += 256) a += logd[i];
  s[threadIdx.x] = a;
  __syncthreads();
  for (int off = 128; off; off >>= 1) {
    if (threadIdx.x < off) s[threadIdx.x] += s[threadIdx.x + off];
    __syncthreads();
  }
  if (threadIdx.x == 0) out[0] = -s[0] / (float)N_ROWS;
}

// ---------------------------------------------------------------------------
// Launch
// ---------------------------------------------------------------------------
extern "C" void kernel_launch(void* const* d_in, const int* in_sizes, int n_in,
                              void* d_out, int out_size, void* d_ws, size_t ws_size,
                              hipStream_t stream) {
  const float* X = (const float*)d_in[0];

  char* ws = (char*)d_ws;
  const size_t bfBytes = (size_t)N_ROWS * DIM * sizeof(unsigned short);
  __bf16* Xhi  = (__bf16*)(ws);
  __bf16* Xlo  = (__bf16*)(ws + bfBytes);
  float*  invn = (float*) (ws + 2 * bfBytes);
  int*    bIdx = (int*)   (ws + 2 * bfBytes + (size_t)N_ROWS * 4);
  float*  logd = (float*) (ws + 2 * bfBytes + (size_t)N_ROWS * 8);

  normalize_split_kernel<<<N_ROWS, 256, 0, stream>>>(X, Xhi, Xlo, invn);
  gram_argmax_kernel<<<N_ROWS / 128, 128, 0, stream>>>(Xhi, Xlo, bIdx);
  logdist_kernel<<<N_ROWS, 256, 0, stream>>>(X, invn, bIdx, logd);
  reduce_loss_kernel<<<1, 256, 0, stream>>>(logd, (float*)d_out);
}
